// ODA2LunaLayer_85341000171934
// MI455X (gfx1250) — matrix operand order repaired and checked by reading, not compiled, not run on hardware
//
#include <hip/hip_runtime.h>

// ---------------------------------------------------------------------------
// CDNA5 (gfx1250) implementation of the Luna-style layer.
// All GEMMs and attention use v_wmma_f32_16x16x32_f16 (wave32).
// ---------------------------------------------------------------------------

typedef _Float16 f16;
typedef __attribute__((ext_vector_type(16))) _Float16 v16h;
typedef __attribute__((ext_vector_type(8)))  _Float16 v8h;
typedef __attribute__((ext_vector_type(8)))  float    v8f;

static __device__ __forceinline__ v8f wmma_ff(v16h a, v16h b, v8f c) {
  // D = A(16x32 f16) * B(32x16 f16) + C(16x16 f32)
  return __builtin_amdgcn_wmma_f32_16x16x32_f16(false, a, false, b, (short)0, c,
                                                false, false);
}

// A-fragment (16x32, f16): lane holds row (lane&15); halves j = K at
// (lane>>4)*16 + j  (16 consecutive halves per lane).
static __device__ __forceinline__ v16h load_afrag(const f16* __restrict__ p,
                                                  size_t stride, int r0, int c0,
                                                  int lane) {
  const f16* q = p + (size_t)(r0 + (lane & 15)) * stride + (size_t)c0
                   + (size_t)((lane >> 4) << 4);
  v8h lo = *(const v8h*)q;
  v8h hi = *(const v8h*)(q + 8);
  return __builtin_shufflevector(lo, hi, 0,1,2,3,4,5,6,7,8,9,10,11,12,13,14,15);
}

// B-fragment (32x16, f16): lane holds column (lane&15); halves j = K at
// (lane>>4)*16 + j  (16 consecutive halves per lane).
static __device__ __forceinline__ v16h load_bfrag(const f16* __restrict__ p,
                                                  size_t stride, int r0, int c0,
                                                  int lane) {
  const f16* q = p + (size_t)(r0 + (lane & 15)) * stride + (size_t)c0
                   + (size_t)((lane >> 4) << 4);
  v8h lo = *(const v8h*)q;
  v8h hi = *(const v8h*)(q + 8);
  return __builtin_shufflevector(lo, hi, 0,1,2,3,4,5,6,7,8,9,10,11,12,13,14,15);
}

static __device__ __forceinline__ float gelu_tanh(float x) {
  // jax.nn.gelu(approximate=True)
  return 0.5f * x * (1.0f + tanhf(0.7978845608028654f * (x + 0.044715f * x * x * x)));
}

static __device__ __forceinline__ unsigned pack2h(float a, float b) {
  union { unsigned u; _Float16 h[2]; } x;
  x.h[0] = (_Float16)a;
  x.h[1] = (_Float16)b;
  return x.u;
}

// ---------------------------------------------------------------------------
// Generic GEMM: C[M,N] = act(A[M,K] * W[N,K]^T + bias[N]).
// One wave per 16x64 output tile. store_t=1 writes C transposed ([N,M]),
// used to produce V^T directly for the attention kernel.
// ---------------------------------------------------------------------------
__global__ __launch_bounds__(32)
void gemm_wmma(const f16* __restrict__ A, const f16* __restrict__ W,
               const float* __restrict__ bias, f16* __restrict__ C,
               int M, int N, int K, int act, int store_t) {
  const int lane = threadIdx.x;
  const int m0 = blockIdx.x * 16;
  const int n0 = blockIdx.y * 64;

  v8f acc0 = {}, acc1 = {}, acc2 = {}, acc3 = {};
  for (int k0 = 0; k0 < K; k0 += 32) {
    v16h a  = load_afrag(A, (size_t)K, m0, k0, lane);
    v16h b0 = load_bfrag(W, (size_t)K, n0 +  0, k0, lane);
    v16h b1 = load_bfrag(W, (size_t)K, n0 + 16, k0, lane);
    v16h b2 = load_bfrag(W, (size_t)K, n0 + 32, k0, lane);
    v16h b3 = load_bfrag(W, (size_t)K, n0 + 48, k0, lane);
    acc0 = wmma_ff(a, b0, acc0);
    acc1 = wmma_ff(a, b1, acc1);
    acc2 = wmma_ff(a, b2, acc2);
    acc3 = wmma_ff(a, b3, acc3);
  }

  const int colb = lane & 15;
  const int rowh = (lane >> 4) * 8;
  v8f accs[4] = {acc0, acc1, acc2, acc3};
#pragma unroll
  for (int t = 0; t < 4; ++t) {
    const int n = n0 + 16 * t + colb;
    const float bv = bias ? bias[n] : 0.0f;
#pragma unroll
    for (int r = 0; r < 8; ++r) {
      float v = accs[t][r] + bv;
      if (act == 1) v = gelu_tanh(v);
      const int m = m0 + rowh + r;
      if (store_t == 0) C[(size_t)m * N + n] = (f16)v;
      else              C[(size_t)n * M + m] = (f16)v;
    }
  }
}

// ---------------------------------------------------------------------------
// Fused flash attention. One wave handles 16 queries for one (batch, head).
//   S^T tile   = K_tile(16x32) x Q^T(32x16)      -> lane = query column
//   online softmax: per-lane + one shfl_xor(16)
//   O^T  accum = V^T_tile(16x32keys) x P^T(32x16) (two hd tiles: 0..15,16..31)
// Q,K: row-major [tokens, 256] (head h at cols h*32..h*32+31)
// VT : row-major [256, Mvt]   (row = h*32+hd, col = batch-token)
// O  : row-major [tokens, 256]
// ---------------------------------------------------------------------------
__global__ __launch_bounds__(32)
void attn_flash(const f16* __restrict__ Q, const f16* __restrict__ Km,
                const f16* __restrict__ VT, f16* __restrict__ O,
                int Lq, int Lk, int Mvt, float scale) {
  const int lane  = threadIdx.x;
  const int q0    = blockIdx.x * 16;
  const int h     = blockIdx.y;
  const int b     = blockIdx.z;
  const int qbase = b * Lq;
  const int kbase = b * Lk;
  const int hc    = h * 32;

  const v16h qf = load_bfrag(Q, 256, qbase + q0, hc, lane);  // Q^T as B-operand

  v8f o0 = {}, o1 = {};              // O^T accumulators: hd 0..15 and 16..31
  float m_run = -1e30f, l_run = 0.0f;

  for (int kc = 0; kc < Lk; kc += 32) {
    const v16h ka0 = load_afrag(Km, 256, kbase + kc,      hc, lane);
    const v16h ka1 = load_afrag(Km, 256, kbase + kc + 16, hc, lane);
    v8f z = {};
    v8f s0 = wmma_ff(ka0, qf, z);    // S^T keys kc..kc+15
    v8f s1 = wmma_ff(ka1, qf, z);    // S^T keys kc+16..kc+31

    float mc = -1e30f;
#pragma unroll
    for (int r = 0; r < 8; ++r) {
      s0[r] *= scale; s1[r] *= scale;
      mc = fmaxf(mc, fmaxf(s0[r], s1[r]));
    }
    mc = fmaxf(mc, __shfl_xor(mc, 16, 32));     // chunk max over all 32 keys
    const float mn = fmaxf(m_run, mc);

    float p0[8], p1[8];
    float lc = 0.0f;
#pragma unroll
    for (int r = 0; r < 8; ++r) {
      p0[r] = __expf(s0[r] - mn);
      p1[r] = __expf(s1[r] - mn);
      lc += p0[r] + p1[r];
    }
    lc += __shfl_xor(lc, 16, 32);
    const float alpha = __expf(m_run - mn);
    l_run = l_run * alpha + lc;
    m_run = mn;
    o0 *= alpha;
    o1 *= alpha;

    // Build P^T B-fragment: lane q (<16) needs keys 0..15 of its query;
    // lane q+16 needs keys 16..31. Own lane holds half of each; swap the
    // other half with the partner lane via shfl_xor(16).
    unsigned own0[4], own1[4], par0[4], par1[4];
#pragma unroll
    for (int j = 0; j < 4; ++j) {
      own0[j] = pack2h(p0[2 * j], p0[2 * j + 1]);
      own1[j] = pack2h(p1[2 * j], p1[2 * j + 1]);
      par0[j] = (unsigned)__shfl_xor((int)own0[j], 16, 32);
      par1[j] = (unsigned)__shfl_xor((int)own1[j], 16, 32);
    }
    const bool low = (lane < 16);
    union { v16h v; unsigned u[8]; } pb;
#pragma unroll
    for (int j = 0; j < 4; ++j) {
      pb.u[j]     = low ? own0[j] : par1[j];
      pb.u[4 + j] = low ? par0[j] : own1[j];
    }

    const v16h vt0 = load_afrag(VT, (size_t)Mvt, hc,      kbase + kc, lane);
    const v16h vt1 = load_afrag(VT, (size_t)Mvt, hc + 16, kbase + kc, lane);
    o0 = wmma_ff(vt0, pb.v, o0);
    o1 = wmma_ff(vt1, pb.v, o1);
  }

  const float rl = 1.0f / l_run;
  const int q    = qbase + q0 + (lane & 15);
  const int hsel = (lane >> 4) * 8;
#pragma unroll
  for (int r = 0; r < 8; ++r) {
    O[(size_t)q * 256 + hc + hsel + r]      = (f16)(o0[r] * rl);
    O[(size_t)q * 256 + hc + 16 + hsel + r] = (f16)(o1[r] * rl);
  }
}

// ---------------------------------------------------------------------------
// LayerNorm(a + o) over dim=256. One 256-thread block per token.
// ---------------------------------------------------------------------------
__global__ __launch_bounds__(256)
void ln_residual(const f16* __restrict__ a, const f16* __restrict__ o,
                 const float* __restrict__ g, const float* __restrict__ bta,
                 f16* __restrict__ outh, float* __restrict__ outf) {
  const int t = blockIdx.x;
  const int i = threadIdx.x;
  const size_t idx = (size_t)t * 256 + i;
  const float v = (float)a[idx] + (float)o[idx];

  __shared__ float sm[8];
  const int lane = i & 31, wid = i >> 5;

  float s = v;
#pragma unroll
  for (int off = 16; off > 0; off >>= 1) s += __shfl_xor(s, off, 32);
  if (lane == 0) sm[wid] = s;
  __syncthreads();
  float tot = 0.0f;
#pragma unroll
  for (int j = 0; j < 8; ++j) tot += sm[j];
  const float mean = tot * (1.0f / 256.0f);
  __syncthreads();

  const float d = v - mean;
  float s2 = d * d;
#pragma unroll
  for (int off = 16; off > 0; off >>= 1) s2 += __shfl_xor(s2, off, 32);
  if (lane == 0) sm[wid] = s2;
  __syncthreads();
  float tv = 0.0f;
#pragma unroll
  for (int j = 0; j < 8; ++j) tv += sm[j];
  const float rstd = rsqrtf(tv * (1.0f / 256.0f) + 1e-5f);

  const float y = d * rstd * g[i] + bta[i];
  outh[idx] = (f16)y;
  if (outf) outf[idx] = y;
}

// --------------------------- small utility kernels -------------------------
__global__ void cvt_f32_f16(const float* __restrict__ in, f16* __restrict__ out,
                            int n) {
  const int i = blockIdx.x * blockDim.x + threadIdx.x;
  if (i < n) out[i] = (f16)in[i];
}

// x [B, C, HW] f32 -> xt [B*HW, C] f16
__global__ void x_to_tokens(const float* __restrict__ x, f16* __restrict__ xt,
                            int Cn, int HW, int n) {
  const int i = blockIdx.x * blockDim.x + threadIdx.x;
  if (i >= n) return;
  const int b = i / (Cn * HW);
  const int r = i % (Cn * HW);
  const int c = r / HW;
  const int p = r % HW;
  xt[((size_t)b * HW + p) * Cn + c] = (f16)x[i];
}

// o [B*HW, OUT] f16 -> wc [B, OUT, HW] f32
__global__ void write_wc(const f16* __restrict__ o, float* __restrict__ out,
                         int Cn, int HW, int n) {
  const int i = blockIdx.x * blockDim.x + threadIdx.x;
  if (i >= n) return;
  const int b = i / (HW * Cn);
  const int r = i % (HW * Cn);
  const int p = r / Cn;
  const int c = r % Cn;
  out[(size_t)b * Cn * HW + (size_t)c * HW + p] = (float)o[i];
}

// ---------------------------------------------------------------------------
// Host-side orchestration.
// Input order (flattened setup_inputs dict):
//  0:x 1:aux | 2..9 q/k/v/o_self (w,b) | 10,11 norm_self(g,b)
//  12..19 q/k/v/o_cross1 (w,b) | 20,21 norm_cross1 | 22..29 q/k/v/o_cross2
//  30,31 ff1 | 32,33 ff2 | 34,35 norm_ff
// ---------------------------------------------------------------------------
extern "C" void kernel_launch(void* const* d_in, const int* in_sizes, int n_in,
                              void* d_out, int out_size, void* d_ws,
                              size_t ws_size, hipStream_t stream) {
  (void)in_sizes; (void)n_in; (void)out_size;
  const int Bn = 8, Cn = 256, HW = 48 * 48, S = 256, D = 256, FF = 1024;
  const int TA = Bn * S;   // 2048 aux tokens (all batches)
  const int TX = Bn * HW;  // 18432 image tokens (all batches)

  const float* x = (const float*)d_in[0];
  const float* aux = (const float*)d_in[1];
  auto F = [&](int i) { return (const float*)d_in[i]; };

  // Workspace layout (f16 units, 256B-aligned blocks).
  f16* base = (f16*)d_ws;
  size_t off = 0;
  auto alloc = [&](size_t n) {
    f16* p = base + off;
    off += (n + 127) & ~(size_t)127;
    return p;
  };
  f16* xth   = alloc((size_t)TX * Cn);   // image tokens f16
  f16* auxA  = alloc((size_t)TA * D);    // aux ping
  f16* auxB  = alloc((size_t)TA * D);    // aux pong
  f16* wbuf  = alloc(12 * (size_t)D * D + 2 * (size_t)D * FF);  // f16 weights
  f16* qbuf  = alloc((size_t)TX * D);
  f16* kbuf  = alloc((size_t)TX * D);
  f16* vtbuf = alloc((size_t)TX * D);    // V^T: [256, tokens]
  f16* attno = alloc((size_t)TX * D);    // attention outputs (concat heads)
  f16* obuf  = alloc((size_t)TA * D);    // o-projection output for aux stages
  f16* ffbuf = alloc((size_t)TA * FF);   // FFN hidden
  if (ws_size < off * sizeof(f16)) return;  // insufficient scratch: no-op

  // --- convert weights to f16 ---
  const int widx[14] = {2, 4, 6, 8, 12, 14, 16, 18, 22, 24, 26, 28, 30, 32};
  f16* wp[14];
  {
    size_t w = 0;
    for (int i = 0; i < 14; ++i) {
      const size_t cnt = (i >= 12) ? (size_t)D * FF : (size_t)D * D;
      wp[i] = wbuf + w;
      w += cnt;
      cvt_f32_f16<<<dim3((unsigned)((cnt + 255) / 256)), 256, 0, stream>>>(
          F(widx[i]), wp[i], (int)cnt);
    }
  }
  // --- convert activations ---
  cvt_f32_f16<<<dim3((TA * D + 255) / 256), 256, 0, stream>>>(aux, auxA, TA * D);
  x_to_tokens<<<dim3((TX * Cn + 255) / 256), 256, 0, stream>>>(x, xth, Cn, HW,
                                                               TX * Cn);

  auto gemm = [&](const f16* A, const f16* W, const float* bias, f16* C, int M,
                  int N, int K, int act, int st) {
    gemm_wmma<<<dim3(M / 16, N / 64), 32, 0, stream>>>(A, W, bias, C, M, N, K,
                                                       act, st);
  };
  auto attn = [&](const f16* Qp, const f16* Kp, const f16* VTp, f16* Op, int Lq,
                  int Lk, int Mvt) {
    attn_flash<<<dim3(Lq / 16, 8, Bn), 32, 0, stream>>>(
        Qp, Kp, VTp, Op, Lq, Lk, Mvt, 0.17677669529663687f /* 1/sqrt(32) */);
  };
  auto ln = [&](const f16* a, const f16* o, const float* g, const float* b,
                f16* oh, float* of) {
    ln_residual<<<dim3(TA), 256, 0, stream>>>(a, o, g, b, oh, of);
  };

  // ---------------- Stage A: self-attention over aux ----------------
  gemm(auxA, wp[0], F(3), qbuf, TA, D, D, 0, 0);      // Q
  gemm(auxA, wp[1], F(5), kbuf, TA, D, D, 0, 0);      // K
  gemm(auxA, wp[2], F(7), vtbuf, TA, D, D, 0, 1);     // V^T [256, TA]
  attn(qbuf, kbuf, vtbuf, attno, S, S, TA);
  gemm(attno, wp[3], F(9), obuf, TA, D, D, 0, 0);     // O proj
  ln(auxA, obuf, F(10), F(11), auxB, nullptr);        // aux1

  // ---------------- Stage B: aux attends to image tokens ----------------
  gemm(auxB, wp[4], F(13), qbuf, TA, D, D, 0, 0);     // Q from aux1
  gemm(xth, wp[5], F(15), kbuf, TX, D, Cn, 0, 0);     // K from xt
  gemm(xth, wp[6], F(17), vtbuf, TX, D, Cn, 0, 1);    // V^T [256, TX]
  attn(qbuf, kbuf, vtbuf, attno, S, HW, TX);
  gemm(attno, wp[7], F(19), obuf, TA, D, D, 0, 0);    // O proj
  ln(auxB, obuf, F(20), F(21), auxA, nullptr);        // aux2

  // ---------------- FFN on aux ----------------
  gemm(auxA, wp[12], F(31), ffbuf, TA, FF, D, 1, 0);  // gelu(ff1)
  gemm(ffbuf, wp[13], F(33), obuf, TA, D, FF, 0, 0);  // ff2
  ln(auxA, obuf, F(34), F(35), auxB, (float*)d_out);  // aux3 (+ f32 output)

  // ---------------- Stage D: image tokens attend to aux ----------------
  gemm(xth, wp[8], F(23), qbuf, TX, D, Cn, 0, 0);     // Q from xt
  gemm(auxB, wp[9], F(25), kbuf, TA, D, D, 0, 0);     // K from aux3
  gemm(auxB, wp[10], F(27), vtbuf, TA, D, D, 0, 1);   // V^T [256, TA]
  attn(qbuf, kbuf, vtbuf, attno, HW, S, TA);
  gemm(attno, wp[11], F(29), kbuf, TX, D, D, 0, 0);   // o_cross2 -> reuse kbuf
  write_wc<<<dim3((TX * Cn + 255) / 256), 256, 0, stream>>>(
      kbuf, (float*)d_out + (size_t)TA * D, Cn, HW, TX * Cn);
}
